// TransformerBlock_70506183131656
// MI455X (gfx1250) — matrix operand order repaired and checked by reading, not compiled, not run on hardware
//
#include <hip/hip_runtime.h>
#include <cstddef>

// ---------------- problem constants ----------------
#define B_ 16
#define R_ 2048
#define U_ 2048
#define D_ 1024

typedef __attribute__((ext_vector_type(16))) __bf16 v16bf;
typedef __attribute__((ext_vector_type(8)))  __bf16 v8bf;
typedef __attribute__((ext_vector_type(4)))  __bf16 v4bf;
typedef __attribute__((ext_vector_type(8)))  float  v8f;
typedef __attribute__((ext_vector_type(4)))  float  v4f;

// ---------------- workspace layout (bytes) ----------------
static constexpr size_t KB_OFF  = 0;                                   // K bf16 [B][U][D]
static constexpr size_t VT_OFF  = KB_OFF  + (size_t)B_*U_*D_*2;        // V^T bf16 [B][D][U]
static constexpr size_t W1T_OFF = VT_OFF  + (size_t)B_*D_*U_*2;        // W1^T bf16 [D][D]
static constexpr size_t W2T_OFF = W1T_OFF + (size_t)D_*D_*2;           // W2^T bf16 [D][D]
static constexpr size_t XF_OFF  = W2T_OFF + (size_t)D_*D_*2;           // X f32 [B*R][D]
static constexpr size_t XB_OFF  = XF_OFF  + (size_t)B_*R_*D_*4;        // X bf16 [B*R][D]
static constexpr size_t HB_OFF  = XB_OFF  + (size_t)B_*R_*D_*2;        // H bf16 [B*R][D]

// A-fragment gather for 16-bit 16x32 A-matrix (ISA 7.12.2):
// lane l: m = l&15, h = l>>4; elems 0..15 at K = 16h+0..15 (contiguous, matches B)
__device__ __forceinline__ v16bf make_afrag(const __bf16* row, int d0, int h) {
  v8bf lo = *(const v8bf*)(row + d0 + 16 * h);
  v8bf hi = *(const v8bf*)(row + d0 + 16 * h + 8);
  return __builtin_shufflevector(lo, hi, 0,1,2,3,4,5,6,7,8,9,10,11,12,13,14,15);
}

__device__ __forceinline__ v8f wmma_bf16(v16bf a, v16bf b, v8f c) {
  return __builtin_amdgcn_wmma_f32_16x16x32_bf16(false, a, false, b, (short)0, c, false, false);
}

// ---------------- prep kernels ----------------
// f32 -> bf16 copy, vectorized: b128 load -> b64 store per thread-iteration.
__global__ __launch_bounds__(256) void cvt_bf16_kernel(const float* __restrict__ src,
                                                       __bf16* __restrict__ dst, size_t n4) {
  size_t stride = (size_t)gridDim.x * blockDim.x;
  for (size_t i = (size_t)blockIdx.x * blockDim.x + threadIdx.x; i < n4; i += stride) {
    v4f v = *(const v4f*)(src + 4 * i);
    v4bf o;
    o.x = (__bf16)v.x; o.y = (__bf16)v.y; o.z = (__bf16)v.z; o.w = (__bf16)v.w;
    *(v4bf*)(dst + 4 * i) = o;
  }
}

// Tiled transpose + convert: src [nb][rows][cols] f32 -> dst [nb][cols][rows] bf16.
// 32x32 tile per block, 256 threads (32 wide x 8 deep, 4 row-iterations).
// LDS tile padded to stride 34 bf16 (17 dwords; gcd(17,64)=1 -> conflict-free column read).
__global__ __launch_bounds__(256) void transpose_cvt_kernel(
    const float* __restrict__ src, __bf16* __restrict__ dst, int rows, int cols) {
  __shared__ __bf16 tile[32 * 34];
  const int tpr = cols >> 5;              // tiles per row of tiles
  const int tpc = rows >> 5;              // tiles per col of tiles
  int bid = blockIdx.x;
  const int ct = bid % tpr; bid /= tpr;   // tile col (along cols)
  const int rt = bid % tpc; bid /= tpc;   // tile row (along rows)
  const int nb = bid;                     // batch index
  const int tx = threadIdx.x & 31;
  const int ty = threadIdx.x >> 5;

  const float* s = src + ((size_t)nb * rows + (size_t)rt * 32) * cols + ct * 32;
#pragma unroll
  for (int r = 0; r < 4; ++r) {
    int row = ty + r * 8;
    tile[row * 34 + tx] = (__bf16)s[(size_t)row * cols + tx];
  }
  __syncthreads();
  __bf16* d = dst + ((size_t)nb * cols + (size_t)ct * 32) * rows + rt * 32;
#pragma unroll
  for (int r = 0; r < 4; ++r) {
    int row = ty + r * 8;                 // output row (a source column)
    d[(size_t)row * rows + tx] = tile[tx * 34 + row];
  }
}

// ---------------- attention + first LayerNorm ----------------
// 1 block = 4 waves = one (batch, 16-row) tile. Dynamic LDS:
//   Qa  bf16[16][1024] @ 0       (32768)
//   Sf  f32 [16][2048] @ 32768   (131072)
//   P   bf16[16][2048] @ 163840  (65536)
//   red1/red2 f32[128] @ 229376 / 229888
//   stat f32[32]       @ 230400
//   Ob  f32 [16][1024] @ 32768   (aliases Sf; used after P is built)
static constexpr unsigned ATTN_LDS = 230528;

__global__ __launch_bounds__(128) void attn_kernel(
    const float* __restrict__ Q, const __bf16* __restrict__ Kb,
    const __bf16* __restrict__ Vt, const float* __restrict__ gamma,
    const float* __restrict__ beta, float* __restrict__ Xf,
    __bf16* __restrict__ Xb)
{
  extern __shared__ char smem[];
  __bf16* Qa  = (__bf16*)(smem);
  float*  Sf  = (float*)(smem + 32768);
  __bf16* P   = (__bf16*)(smem + 163840);
  float* red1 = (float*)(smem + 229376);
  float* red2 = (float*)(smem + 229888);
  float* stat = (float*)(smem + 230400);
  float* Ob   = (float*)(smem + 32768);

  const int tid  = threadIdx.x;
  const int wave = tid >> 5;
  const int lane = tid & 31;
  const int m16  = lane & 15;
  const int h    = lane >> 4;
  const int bat  = blockIdx.x / (R_ / 16);
  const int r0   = (blockIdx.x % (R_ / 16)) * 16;

  // ---- phase 1: Q tile -> bf16 LDS ----
  const float* Qtile = Q + ((size_t)bat * R_ + r0) * D_;
  for (int i = tid; i < 16 * D_; i += 128)
    Qa[i] = (__bf16)Qtile[i];
  __syncthreads();

  // ---- phase 2: scores S = Q K^T / sqrt(D+eps) ----
  const float scale = rsqrtf((float)D_ + 1e-8f);
  const __bf16* Kbase = Kb + (size_t)bat * U_ * D_;
  for (int jj = 0; jj < 8; ++jj) {
    int nt0 = wave * 32 + jj * 4;   // this wave owns keys [wave*512, wave*512+512)
    v8f acc[4] = {};
    for (int kc = 0; kc < D_ / 32; ++kc) {
      int d0 = kc * 32;
      v16bf a = make_afrag(Qa + m16 * D_, d0, h);
#pragma unroll
      for (int t = 0; t < 4; ++t) {
        const __bf16* bp = Kbase + ((size_t)(nt0 + t) * 16 + m16) * D_ + d0 + 16 * h;
        v16bf bf = *(const v16bf*)bp;
        acc[t] = wmma_bf16(a, bf, acc[t]);
      }
    }
#pragma unroll
    for (int t = 0; t < 4; ++t) {
      int col = (nt0 + t) * 16 + m16;
#pragma unroll
      for (int r = 0; r < 8; ++r)
        Sf[(r + 8 * h) * U_ + col] = acc[t][r] * scale;
    }
  }
  __syncthreads();

  // ---- phase 2b: softmax rows (16 rows x 2048), 8 threads per row ----
  {
    const int r = tid >> 3, c = tid & 7;
    const float* srow = Sf + r * U_;
    float lmax = -3.4e38f;
    for (int u = c * 256; u < c * 256 + 256; ++u) lmax = fmaxf(lmax, srow[u]);
    red1[r * 8 + c] = lmax;
    __syncthreads();
    if (c == 0) {
      float m = red1[r * 8];
      for (int i = 1; i < 8; ++i) m = fmaxf(m, red1[r * 8 + i]);
      stat[r] = m;
    }
    __syncthreads();
    const float rm = stat[r];
    float lsum = 0.f;
    for (int u = c * 256; u < c * 256 + 256; ++u) lsum += __expf(srow[u] - rm);
    red2[r * 8 + c] = lsum;
    __syncthreads();
    if (c == 0) {
      float s = 0.f;
      for (int i = 0; i < 8; ++i) s += red2[r * 8 + i];
      stat[16 + r] = 1.0f / s;
    }
    __syncthreads();
    const float inv = stat[16 + r];
    __bf16* prow = P + r * U_;
    for (int u = c * 256; u < c * 256 + 256; ++u)
      prow[u] = (__bf16)(__expf(srow[u] - rm) * inv);
  }
  __syncthreads();

  // ---- phase 3: O = P @ V  (V pre-transposed [D][U]) ----
  const __bf16* Vbase = Vt + (size_t)bat * D_ * U_;
  for (int jj = 0; jj < 4; ++jj) {
    int nt0 = wave * 16 + jj * 4;   // this wave owns cols [wave*256, +256)
    v8f acc[4] = {};
    for (int kc = 0; kc < U_ / 32; ++kc) {
      int u0 = kc * 32;
      v16bf a = make_afrag(P + m16 * U_, u0, h);
#pragma unroll
      for (int t = 0; t < 4; ++t) {
        const __bf16* bp = Vbase + ((size_t)(nt0 + t) * 16 + m16) * U_ + u0 + 16 * h;
        v16bf bf = *(const v16bf*)bp;
        acc[t] = wmma_bf16(a, bf, acc[t]);
      }
    }
#pragma unroll
    for (int t = 0; t < 4; ++t) {
      int col = (nt0 + t) * 16 + m16;
#pragma unroll
      for (int r = 0; r < 8; ++r)
        Ob[(r + 8 * h) * D_ + col] = acc[t][r];
    }
  }
  __syncthreads();

  // ---- phase 4: X = LayerNorm(Q + O), write f32 + bf16 ----
  {
    const int r = tid >> 3, c = tid & 7;
    const float* qrow = Q + ((size_t)bat * R_ + r0 + r) * D_;
    const float* orow = Ob + r * D_;
    float s = 0.f, s2 = 0.f;
    for (int d = c * 128; d < c * 128 + 128; ++d) {
      float x = qrow[d] + orow[d];
      s += x; s2 += x * x;
    }
    red1[r * 8 + c] = s; red2[r * 8 + c] = s2;
    __syncthreads();
    if (c == 0) {
      float S = 0.f, S2 = 0.f;
      for (int i = 0; i < 8; ++i) { S += red1[r * 8 + i]; S2 += red2[r * 8 + i]; }
      float mu = S / (float)D_;
      float var = S2 / (float)D_ - mu * mu;
      stat[r] = mu;
      stat[16 + r] = rsqrtf(var + 1e-6f);
    }
    __syncthreads();
    const float mu = stat[r], rs = stat[16 + r];
    float* xrow = Xf + ((size_t)bat * R_ + r0 + r) * D_;
    __bf16* xbrow = Xb + ((size_t)bat * R_ + r0 + r) * D_;
    for (int d = c * 128; d < c * 128 + 128; ++d) {
      float x = qrow[d] + orow[d];
      float y = gamma[d] * (x - mu) * rs + beta[d];
      xrow[d] = y;
      xbrow[d] = (__bf16)y;
    }
  }
}

// ---------------- FFN layer 1: H = relu(X W1 + b1) ----------------
__global__ __launch_bounds__(128) void ffn1_kernel(
    const __bf16* __restrict__ Xb, const __bf16* __restrict__ W1t,
    const float* __restrict__ b1, __bf16* __restrict__ Hb)
{
  extern __shared__ char smem[];
  __bf16* Xa = (__bf16*)smem;   // [16][1024]

  const int tid = threadIdx.x;
  const int wave = tid >> 5, lane = tid & 31;
  const int m16 = lane & 15, h = lane >> 4;
  const size_t row0 = (size_t)blockIdx.x * 16;

  const __bf16* xtile = Xb + row0 * D_;
  for (int i = tid; i < 16 * D_; i += 128) Xa[i] = xtile[i];
  __syncthreads();

  for (int jj = 0; jj < 4; ++jj) {
    int nt0 = wave * 16 + jj * 4;
    v8f acc[4] = {};
    for (int kc = 0; kc < D_ / 32; ++kc) {
      int k0 = kc * 32;
      v16bf a = make_afrag(Xa + m16 * D_, k0, h);
#pragma unroll
      for (int t = 0; t < 4; ++t) {
        const __bf16* bp = W1t + ((size_t)(nt0 + t) * 16 + m16) * D_ + k0 + 16 * h;
        v16bf bf = *(const v16bf*)bp;
        acc[t] = wmma_bf16(a, bf, acc[t]);
      }
    }
#pragma unroll
    for (int t = 0; t < 4; ++t) {
      int col = (nt0 + t) * 16 + m16;
      float bias = b1[col];
#pragma unroll
      for (int r = 0; r < 8; ++r) {
        float v = fmaxf(acc[t][r] + bias, 0.0f);
        Hb[(row0 + r + 8 * h) * D_ + col] = (__bf16)v;
      }
    }
  }
}

// ---------------- FFN layer 2 + residual + second LayerNorm ----------------
static constexpr unsigned FFN2_LDS = 99456;

__global__ __launch_bounds__(128) void ffn2_kernel(
    const __bf16* __restrict__ Hb, const __bf16* __restrict__ W2t,
    const float* __restrict__ b2, const float* __restrict__ Xf,
    const float* __restrict__ gamma, const float* __restrict__ beta,
    float* __restrict__ Out)
{
  extern __shared__ char smem[];
  __bf16* Ha  = (__bf16*)smem;            // [16][1024]
  float*  Ob  = (float*)(smem + 32768);   // [16][1024]
  float* red1 = (float*)(smem + 98304);
  float* red2 = (float*)(smem + 98816);
  float* stat = (float*)(smem + 99328);

  const int tid = threadIdx.x;
  const int wave = tid >> 5, lane = tid & 31;
  const int m16 = lane & 15, h = lane >> 4;
  const size_t row0 = (size_t)blockIdx.x * 16;

  const __bf16* htile = Hb + row0 * D_;
  for (int i = tid; i < 16 * D_; i += 128) Ha[i] = htile[i];
  __syncthreads();

  for (int jj = 0; jj < 4; ++jj) {
    int nt0 = wave * 16 + jj * 4;
    v8f acc[4] = {};
    for (int kc = 0; kc < D_ / 32; ++kc) {
      int k0 = kc * 32;
      v16bf a = make_afrag(Ha + m16 * D_, k0, h);
#pragma unroll
      for (int t = 0; t < 4; ++t) {
        const __bf16* bp = W2t + ((size_t)(nt0 + t) * 16 + m16) * D_ + k0 + 16 * h;
        v16bf bf = *(const v16bf*)bp;
        acc[t] = wmma_bf16(a, bf, acc[t]);
      }
    }
#pragma unroll
    for (int t = 0; t < 4; ++t) {
      int col = (nt0 + t) * 16 + m16;
      float bias = b2[col];
#pragma unroll
      for (int r = 0; r < 8; ++r) {
        size_t grow = row0 + r + 8 * h;
        Ob[(r + 8 * h) * D_ + col] = acc[t][r] + bias + Xf[grow * D_ + col];
      }
    }
  }
  __syncthreads();

  // LayerNorm over Ob rows, write final output
  {
    const int r = tid >> 3, c = tid & 7;
    const float* zrow = Ob + r * D_;
    float s = 0.f, s2 = 0.f;
    for (int d = c * 128; d < c * 128 + 128; ++d) { float x = zrow[d]; s += x; s2 += x * x; }
    red1[r * 8 + c] = s; red2[r * 8 + c] = s2;
    __syncthreads();
    if (c == 0) {
      float S = 0.f, S2 = 0.f;
      for (int i = 0; i < 8; ++i) { S += red1[r * 8 + i]; S2 += red2[r * 8 + i]; }
      float mu = S / (float)D_;
      float var = S2 / (float)D_ - mu * mu;
      stat[r] = mu;
      stat[16 + r] = rsqrtf(var + 1e-6f);
    }
    __syncthreads();
    const float mu = stat[r], rs = stat[16 + r];
    float* orow = Out + (row0 + r) * D_;
    for (int d = c * 128; d < c * 128 + 128; ++d)
      orow[d] = gamma[d] * (zrow[d] - mu) * rs + beta[d];
  }
}

// ---------------- host launcher ----------------
extern "C" void kernel_launch(void* const* d_in, const int* in_sizes, int n_in,
                              void* d_out, int out_size, void* d_ws, size_t ws_size,
                              hipStream_t stream) {
  const float* Q     = (const float*)d_in[0];
  const float* K     = (const float*)d_in[1];
  const float* V     = (const float*)d_in[2];
  const float* W1    = (const float*)d_in[3];
  const float* b1    = (const float*)d_in[4];
  const float* W2    = (const float*)d_in[5];
  const float* b2    = (const float*)d_in[6];
  const float* gamma = (const float*)d_in[7];
  const float* beta  = (const float*)d_in[8];
  float* Out = (float*)d_out;

  char* ws = (char*)d_ws;
  __bf16* Kb  = (__bf16*)(ws + KB_OFF);
  __bf16* Vt  = (__bf16*)(ws + VT_OFF);
  __bf16* W1t = (__bf16*)(ws + W1T_OFF);
  __bf16* W2t = (__bf16*)(ws + W2T_OFF);
  float*  Xf  = (float*)(ws + XF_OFF);
  __bf16* Xb  = (__bf16*)(ws + XB_OFF);
  __bf16* Hb  = (__bf16*)(ws + HB_OFF);

  const size_t nKV = (size_t)B_ * U_ * D_;
  cvt_bf16_kernel<<<4096, 256, 0, stream>>>(K, Kb, nKV / 4);

  // V: [B][U][D] -> [B][D][U]; tiles = B * (U/32) * (D/32)
  transpose_cvt_kernel<<<B_ * (U_ / 32) * (D_ / 32), 256, 0, stream>>>(V, Vt, U_, D_);
  // W1/W2: [D][D] -> [D][D] transposed
  transpose_cvt_kernel<<<(D_ / 32) * (D_ / 32), 256, 0, stream>>>(W1, W1t, D_, D_);
  transpose_cvt_kernel<<<(D_ / 32) * (D_ / 32), 256, 0, stream>>>(W2, W2t, D_, D_);

  attn_kernel<<<B_ * (R_ / 16), 128, ATTN_LDS, stream>>>(Q, Kb, Vt, gamma, beta, Xf, Xb);
  ffn1_kernel<<<(B_ * R_) / 16, 128, 32768, stream>>>(Xb, W1t, b1, Hb);
  ffn2_kernel<<<(B_ * R_) / 16, 128, FFN2_LDS, stream>>>(Hb, W2t, b2, Xf, gamma, beta, Out);
}
